// Qwen3Attention_37769942401658
// MI455X (gfx1250) — compile-verified, matrix operand order checked by reading, not run on hardware
//
#include <hip/hip_runtime.h>
#include <cstdint>
#include <cstddef>

// ---------------------------------------------------------------------------
// Qwen3 attention block (B=2,S=2048,HID=2048,H=16,KV=8,HD=128,G=2,WIN=128)
// MI455X / gfx1250: all matmuls via v_wmma_f32_16x16x32_bf16 (fp32 accum),
// K/V tiles staged with global_load_async_to_lds_b128 (ASYNCcnt path),
// softmax reductions on the VALU via v_permlane16_b32.
// ---------------------------------------------------------------------------

typedef __attribute__((ext_vector_type(16))) __bf16 v16bf;
typedef __attribute__((ext_vector_type(8)))  __bf16 v8bf;
typedef __attribute__((ext_vector_type(4)))  __bf16 v4bf;
typedef __attribute__((ext_vector_type(8)))  float  v8f;

union V16U { v16bf v; v8bf h[2]; __bf16 e[16]; };
union V4U  { v4bf  v; __bf16 e[4]; };

constexpr int Bc = 2, Sc = 2048, HIDc = 2048;
constexpr int Hc = 16, KVc = 8, HDc = 128, Gc = 2, WINc = 128;
constexpr float ALPHA_SCALEc = 4.0f, ALPHA_OFFSETc = 2.0f, EPSc = 1e-6f;
constexpr float SCALEc = 0.08838834764831845f;   // 128^-0.5
constexpr float NEGc = -1e30f;

// GEMM tiling
constexpr int BMg = 64, BNg = 128, BKg = 32, LDg = 40; // LDS row stride (bf16), padded

// Native f32 -> bf16 (hardware cvt on gfx1250, RNE).
static __device__ __forceinline__ __bf16 f2bf(float f) { return (__bf16)f; }

static __device__ __forceinline__ v8f wmma_bf16(v16bf a, v16bf b, v8f c) {
  return __builtin_amdgcn_wmma_f32_16x16x32_bf16(false, a, false, b,
                                                 (short)0, c, false, false);
}

// --- xor-mask crosslane within rows of 16, on the VALU ----------------------
constexpr unsigned sel_lo(int m) {
  unsigned v = 0;
  for (int i = 0; i < 8; ++i) v |= ((unsigned)((i ^ m) & 0xF)) << (4 * i);
  return v;
}
constexpr unsigned sel_hi(int m) {
  unsigned v = 0;
  for (int i = 8; i < 16; ++i) v |= ((unsigned)((i ^ m) & 0xF)) << (4 * (i - 8));
  return v;
}
template<int M>
static __device__ __forceinline__ float xor16(float x) {
#if __has_builtin(__builtin_amdgcn_permlane16)
  unsigned u = __builtin_bit_cast(unsigned, x);
  unsigned r = __builtin_amdgcn_permlane16(u, u, sel_lo(M), sel_hi(M),
                                           false, false);
  return __builtin_bit_cast(float, r);
#else
  return __shfl_xor(x, M, 16);
#endif
}

// --- CDNA5 async global->LDS copies (tracked by ASYNCcnt) -------------------
static __device__ __forceinline__ unsigned lds_off32(const void* p) {
  return (unsigned)(uintptr_t)p;             // low 32 bits = LDS byte offset
}
static __device__ __forceinline__ void async_load_b128(unsigned lds, const void* g) {
  asm volatile("global_load_async_to_lds_b128 %0, %1, off"
               :: "v"(lds), "v"(g) : "memory");
}
static __device__ __forceinline__ void async_load_b32(unsigned lds, const void* g) {
  asm volatile("global_load_async_to_lds_b32 %0, %1, off"
               :: "v"(lds), "v"(g) : "memory");
}
static __device__ __forceinline__ void wait_async0() {
  asm volatile("s_wait_asynccnt 0x0" ::: "memory");
}

// ---------------------------------------------------------------------------
// Generic GEMM: C[M,N] = A[M,K] * W[N,K]^T  (A,W,C fp32; bf16 WMMA compute).
// N,K compile-time so strides fold to shifts / immediate store offsets.
// Block: 256 threads = 8 waves, tile 64(M) x 128(N), K step 32.
// ---------------------------------------------------------------------------
template<int N, int K>
__global__ __launch_bounds__(256) void gemm_xw_bf16(
    const float* __restrict__ A, const float* __restrict__ W,
    float* __restrict__ C)
{
  __shared__ __align__(16) __bf16 As[BMg * LDg];
  __shared__ __align__(16) __bf16 Bs[BNg * LDg];

  const int tid  = threadIdx.x;
  const int lane = tid & 31, wave = tid >> 5;
  const int half = lane >> 4, l16 = lane & 15;
  const int wm = wave & 1, wn = wave >> 1;
  const int m0 = blockIdx.y * BMg, n0 = blockIdx.x * BNg;

  v8f acc[2][2];
  #pragma unroll
  for (int mi = 0; mi < 2; ++mi)
    #pragma unroll
    for (int ni = 0; ni < 2; ++ni)
      #pragma unroll
      for (int r = 0; r < 8; ++r) acc[mi][ni][r] = 0.f;

  const int ar = tid >> 2, ac = (tid & 3) << 3;  // A: 64 rows x 32 cols, 8/thread
  const int br = tid >> 1, bc = (tid & 1) << 4;  // W: 128 rows x 32 cols, 16/thread
  const float* aptr = A + (size_t)(m0 + ar) * K + ac;
  const float* wptr = W + (size_t)(n0 + br) * K + bc;

  for (int k0 = 0; k0 < K; k0 += BKg, aptr += BKg, wptr += BKg) {
    {
      float4 f0 = ((const float4*)aptr)[0];
      float4 f1 = ((const float4*)aptr)[1];
      __bf16* d = &As[ar * LDg + ac];
      d[0]=f2bf(f0.x); d[1]=f2bf(f0.y); d[2]=f2bf(f0.z); d[3]=f2bf(f0.w);
      d[4]=f2bf(f1.x); d[5]=f2bf(f1.y); d[6]=f2bf(f1.z); d[7]=f2bf(f1.w);
    }
    {
      float4 f0 = ((const float4*)wptr)[0];
      float4 f1 = ((const float4*)wptr)[1];
      float4 f2 = ((const float4*)wptr)[2];
      float4 f3 = ((const float4*)wptr)[3];
      __bf16* d = &Bs[br * LDg + bc];
      d[0]=f2bf(f0.x);  d[1]=f2bf(f0.y);  d[2]=f2bf(f0.z);  d[3]=f2bf(f0.w);
      d[4]=f2bf(f1.x);  d[5]=f2bf(f1.y);  d[6]=f2bf(f1.z);  d[7]=f2bf(f1.w);
      d[8]=f2bf(f2.x);  d[9]=f2bf(f2.y);  d[10]=f2bf(f2.z); d[11]=f2bf(f2.w);
      d[12]=f2bf(f3.x); d[13]=f2bf(f3.y); d[14]=f2bf(f3.z); d[15]=f2bf(f3.w);
    }
    __syncthreads();

    V16U af[2], bf_[2];
    #pragma unroll
    for (int mi = 0; mi < 2; ++mi) {
      const __bf16* p = &As[(wm * 32 + mi * 16 + l16) * LDg];
      af[mi].h[0] = *(const v8bf*)(p + half * 8);
      af[mi].h[1] = *(const v8bf*)(p + 16 + half * 8);
    }
    #pragma unroll
    for (int ni = 0; ni < 2; ++ni) {
      const __bf16* p = &Bs[(wn * 32 + ni * 16 + l16) * LDg];
      bf_[ni].h[0] = *(const v8bf*)(p + half * 8);
      bf_[ni].h[1] = *(const v8bf*)(p + 16 + half * 8);
    }
    #pragma unroll
    for (int mi = 0; mi < 2; ++mi)
      #pragma unroll
      for (int ni = 0; ni < 2; ++ni)
        acc[mi][ni] = wmma_bf16(af[mi].v, bf_[ni].v, acc[mi][ni]);

    __syncthreads();
  }

  #pragma unroll
  for (int mi = 0; mi < 2; ++mi)
    #pragma unroll
    for (int ni = 0; ni < 2; ++ni) {
      const int rb = m0 + wm * 32 + mi * 16 + half * 8;
      const int cc = n0 + wn * 32 + ni * 16 + l16;
      float* cp = C + (size_t)rb * N + cc;
      #pragma unroll
      for (int r = 0; r < 8; ++r) cp[(size_t)r * N] = acc[mi][ni][r];
    }
}

// ---------------------------------------------------------------------------
// RMS-norm + DMS decision + RoPE + bf16 pack. One wave per (b,s,head-slot).
// Row slots per (b,s): 0..15 = q heads, 16..23 = k heads, 24..31 = v heads.
// Q is pre-scaled by 1/sqrt(HD); V is emitted TRANSPOSED vbf[(b,kv,hd,s)].
// ---------------------------------------------------------------------------
__global__ __launch_bounds__(256) void norm_rope_kernel(
    const float* __restrict__ Qf, const float* __restrict__ Kf,
    const float* __restrict__ Vf,
    const float* __restrict__ cosT, const float* __restrict__ sinT,
    const float* __restrict__ qn_w, const float* __restrict__ kn_w,
    __bf16* __restrict__ qbf, __bf16* __restrict__ kbf,
    __bf16* __restrict__ vbfT, float* __restrict__ dec)
{
  const int row  = blockIdx.x * 8 + (threadIdx.x >> 5);
  const int lane = threadIdx.x & 31;
  const int bs   = row >> 5;           // b*S + s
  const int sub  = row & 31;
  const int b = bs / Sc, s = bs % Sc;
  const int d0 = lane << 2;

  if (sub >= 24) {                     // V head: bf16 convert + transpose store
    const int kvh = sub - 24;
    float4 xv = *(const float4*)(Vf + (size_t)bs * (KVc * HDc) + kvh * HDc + d0);
    __bf16* vd = vbfT + ((size_t)(b * KVc + kvh)) * HDc * Sc + s;
    vd[(size_t)(d0 + 0) * Sc] = f2bf(xv.x);
    vd[(size_t)(d0 + 1) * Sc] = f2bf(xv.y);
    vd[(size_t)(d0 + 2) * Sc] = f2bf(xv.z);
    vd[(size_t)(d0 + 3) * Sc] = f2bf(xv.w);
    return;
  }

  const bool isQ  = (sub < 16);
  const int  head = isQ ? sub : sub - 16;
  const float* src = isQ ? (Qf + (size_t)bs * (Hc * HDc)  + head * HDc + d0)
                         : (Kf + (size_t)bs * (KVc * HDc) + head * HDc + d0);
  const float* wgt = isQ ? qn_w : kn_w;

  float4 xv = *(const float4*)src;
  float x[4] = { xv.x, xv.y, xv.z, xv.w };
  float ss = x[0]*x[0] + x[1]*x[1] + x[2]*x[2] + x[3]*x[3];
  #pragma unroll
  for (int o = 16; o >= 1; o >>= 1) ss += __shfl_xor(ss, o, 32);
  const float rn = rsqrtf(ss * (1.0f / HDc) + EPSc);

  float xn[4];
  #pragma unroll
  for (int i = 0; i < 4; ++i) xn[i] = x[i] * rn * wgt[d0 + i];

  const bool decHead = isQ && ((head & (Gc - 1)) == 0);
  if (decHead && lane == 31) {         // channel 127: decision logit (pre-RoPE)
    float logit = xn[3] * ALPHA_SCALEc - ALPHA_OFFSETc;
    dec[((size_t)(b * KVc + (head >> 1))) * Sc + s] = (logit > 0.f) ? 1.f : 0.f;
    xn[3] = 0.f;                       // reference zeroes before RoPE
  }

  const float* cs = cosT + (size_t)s * HDc + d0;
  const float* sn = sinT + (size_t)s * HDc + d0;
  const float qsc = isQ ? SCALEc : 1.0f;    // fold 1/sqrt(HD) into Q
  float y[4];
  #pragma unroll
  for (int i = 0; i < 4; ++i) {        // rotate-half partner lives in lane^16
    float other = __shfl_xor(xn[i], 16, 32);
    float rot = (lane < 16) ? -other : other;
    y[i] = (xn[i] * cs[i] + rot * sn[i]) * qsc;
  }
  if (decHead && lane == 31) y[3] = 0.f;   // reference re-zeroes after RoPE

  V4U o;
  #pragma unroll
  for (int i = 0; i < 4; ++i) o.e[i] = f2bf(y[i]);
  __bf16* dst = isQ ? (qbf + (((size_t)(b * Hc  + head)) * Sc + s) * HDc + d0)
                    : (kbf + (((size_t)(b * KVc + head)) * Sc + s) * HDc + d0);
  *(v4bf*)dst = o.v;
}

// ---------------------------------------------------------------------------
// Flash attention with DMS mask. 8 waves/block; wave owns 16 queries.
// K chunk: [32 keys][128 dims]; V chunk: [128 dims][32 keys] (pre-transposed).
// Both staged by the async global->LDS engine. All WMMA fragments are pairs
// of 16-byte ds_load_b128 reads. QK^T = 8 WMMAs/chunk, PV = 8 WMMAs/chunk.
// ---------------------------------------------------------------------------
__global__ __launch_bounds__(256) void attn_kernel(
    const __bf16* __restrict__ qbf, const __bf16* __restrict__ kbf,
    const __bf16* __restrict__ vbfT, const float* __restrict__ dec,
    float* __restrict__ attn)
{
  constexpr int KLD = HDc + 8;       // K tile row stride (bf16)
  constexpr int VLD = 40;            // V^T tile row stride (bf16)
  constexpr int PLD = 40;
  constexpr int TQ  = Sc / 128;

  __shared__ __align__(16) __bf16 Ks [32 * KLD];
  __shared__ __align__(16) __bf16 VsT[HDc * VLD];     // [hd][key]
  __shared__ __align__(16) __bf16 Ps [8 * 16 * PLD];  // per-wave P staging
  __shared__ __align__(16) float decs[32];

  const int bx   = blockIdx.x;
  const int tile = bx % TQ;
  const int hh   = bx / TQ;
  const int h    = hh % Hc;
  const int b    = hh / Hc;
  const int kv   = h / Gc;

  const int tid  = threadIdx.x;
  const int lane = tid & 31, wave = tid >> 5;
  const int half = lane >> 4, l16 = lane & 15;
  const int q0   = tile * 128 + wave * 16;

  // Preload Q fragments (rows fixed for the whole key sweep): 4 K-chunks of 32.
  V16U qf[4];
  const __bf16* qrow = qbf + (((size_t)(b * Hc + h)) * Sc + (q0 + l16)) * HDc;
  #pragma unroll
  for (int kk = 0; kk < 4; ++kk) {
    qf[kk].h[0] = *(const v8bf*)(qrow + kk * 32 + half * 8);
    qf[kk].h[1] = *(const v8bf*)(qrow + kk * 32 + 16 + half * 8);
  }

  float m_r[8], l_r[8];
  v8f oacc[8];
  #pragma unroll
  for (int r = 0; r < 8; ++r) { m_r[r] = -3.0e38f; l_r[r] = 0.f; }
  #pragma unroll
  for (int t = 0; t < 8; ++t)
    #pragma unroll
    for (int r = 0; r < 8; ++r) oacc[t][r] = 0.f;

  // K staging: thread -> key row tid/8, 16 bf16 at col (tid%8)*16
  const int kr_ = tid >> 3, kc_ = (tid & 7) << 4;
  // V^T staging: thread -> hd row tid/2, 16 keys at col (tid%2)*16
  const int vr_ = tid >> 1, vc_ = (tid & 1) << 4;
  const size_t kvbase = ((size_t)(b * KVc + kv)) * Sc;

  // Loop-carried async source pointers (no per-iteration 64-bit rebuilds).
  const __bf16* ksrc = kbf + (kvbase + kr_) * HDc + kc_;
  const __bf16* vsrc = vbfT + ((size_t)(b * KVc + kv)) * HDc * Sc
                            + (size_t)vr_ * Sc + vc_;
  const float*  dsrc = dec + kvbase + tid;
  __bf16* pw = &Ps[wave * 16 * PLD];

  const unsigned ldsK0 = lds_off32(&Ks [kr_ * KLD + kc_]);
  const unsigned ldsV0 = lds_off32(&VsT[vr_ * VLD + vc_]);
  const unsigned ldsD  = lds_off32(&decs[tid & 31]);

  const int nchunks = tile * 4 + 4;   // keys up to block's max query index
  for (int c = 0; c < nchunks; ++c,
       ksrc += 32 * HDc, vsrc += 32, dsrc += 32) {
    const int k0 = c * 32;
    __syncthreads();                   // prior chunk's LDS reads done
    {
      async_load_b128(ldsK0,      ksrc);
      async_load_b128(ldsK0 + 16, ksrc + 8);
      async_load_b128(ldsV0,      vsrc);
      async_load_b128(ldsV0 + 16, vsrc + 8);
      if (tid < 32) async_load_b32(ldsD, dsrc);
    }
    wait_async0();                     // this wave's async copies landed in LDS
    __syncthreads();                   // all waves' copies visible

    // S = Q * K^T for two 16-key tiles (scale pre-folded into Q)
    v8f sacc[2];
    #pragma unroll
    for (int t = 0; t < 2; ++t)
      #pragma unroll
      for (int r = 0; r < 8; ++r) sacc[t][r] = 0.f;
    #pragma unroll
    for (int t = 0; t < 2; ++t)
      #pragma unroll
      for (int kk = 0; kk < 4; ++kk) {
        V16U bf_;
        const __bf16* kr = &Ks[(t * 16 + l16) * KLD + kk * 32];
        bf_.h[0] = *(const v8bf*)(kr + half * 8);
        bf_.h[1] = *(const v8bf*)(kr + 16 + half * 8);
        sacc[t] = wmma_bf16(qf[kk].v, bf_.v, sacc[t]);
      }

    // Mask + online softmax (rows live in half-wave lane groups), emit P to LDS
    #pragma unroll
    for (int r = 0; r < 8; ++r) {
      const int i = q0 + r + 8 * half;
      float s0 = sacc[0][r];
      float s1 = sacc[1][r];
      {
        const int j0 = k0 + l16;
        const int j1 = k0 + 16 + l16;
        bool a0 = (j0 <= i) && (((i - j0) < WINc) || (decs[l16] == 0.f));
        bool a1 = (j1 <= i) && (((i - j1) < WINc) || (decs[16 + l16] == 0.f));
        if (!a0) s0 = NEGc;
        if (!a1) s1 = NEGc;
      }
      float mx = fmaxf(s0, s1);
      mx = fmaxf(mx, xor16<1>(mx));
      mx = fmaxf(mx, xor16<2>(mx));
      mx = fmaxf(mx, xor16<4>(mx));
      mx = fmaxf(mx, xor16<8>(mx));
      const float mn = fmaxf(m_r[r], mx);
      const float so = __expf(m_r[r] - mn);
      const float p0 = __expf(s0 - mn);
      const float p1 = __expf(s1 - mn);
      float ps = p0 + p1;
      ps += xor16<1>(ps);
      ps += xor16<2>(ps);
      ps += xor16<4>(ps);
      ps += xor16<8>(ps);
      l_r[r] = l_r[r] * so + ps;
      m_r[r] = mn;
      #pragma unroll
      for (int t = 0; t < 8; ++t) oacc[t][r] *= so;
      __bf16* pd = pw + (r + 8 * half) * PLD + l16;
      pd[0]  = f2bf(p0);
      pd[16] = f2bf(p1);
    }
    // Per-wave LDS ordering: make P stores visible before fragment reload.
    asm volatile("s_wait_dscnt 0x0" ::: "memory");

    // Re-load P in A-fragment layout, then O += P * V
    V16U pa;
    const __bf16* pr = pw + l16 * PLD;
    pa.h[0] = *(const v8bf*)(pr + half * 8);
    pa.h[1] = *(const v8bf*)(pr + 16 + half * 8);

    #pragma unroll
    for (int tn = 0; tn < 8; ++tn) {
      V16U vb;
      const __bf16* vrow = &VsT[(tn * 16 + l16) * VLD];
      vb.h[0] = *(const v8bf*)(vrow + half * 8);
      vb.h[1] = *(const v8bf*)(vrow + 16 + half * 8);
      oacc[tn] = wmma_bf16(pa.v, vb.v, oacc[tn]);
    }
  }

  // Epilogue: O /= l, store fp32 in (b*S+i, h*HD+hd) layout for the out-GEMM.
  #pragma unroll
  for (int r = 0; r < 8; ++r) {
    const int i = q0 + r + 8 * half;
    const float inv = 1.0f / l_r[r];
    float* op = attn + ((size_t)(b * Sc) + i) * (Hc * HDc) + h * HDc + l16;
    #pragma unroll
    for (int tn = 0; tn < 8; ++tn) op[tn * 16] = oacc[tn][r] * inv;
  }
}

// ---------------------------------------------------------------------------
extern "C" void kernel_launch(void* const* d_in, const int* in_sizes, int n_in,
                              void* d_out, int out_size, void* d_ws, size_t ws_size,
                              hipStream_t stream)
{
  (void)in_sizes; (void)n_in; (void)out_size; (void)ws_size;
  const float* x    = (const float*)d_in[0];
  const float* cosT = (const float*)d_in[1];
  const float* sinT = (const float*)d_in[2];
  const float* Wq   = (const float*)d_in[3];
  const float* Wk   = (const float*)d_in[4];
  const float* Wv   = (const float*)d_in[5];
  const float* Wo   = (const float*)d_in[6];
  const float* qn_w = (const float*)d_in[7];
  const float* kn_w = (const float*)d_in[8];
  float* out = (float*)d_out;

  char* ws = (char*)d_ws;
  size_t off = 0;
  auto alloc = [&](size_t bytes) -> void* {
    off = (off + 255) & ~(size_t)255;
    void* p = ws + off;
    off += bytes;
    return p;
  };
  const size_t M = (size_t)Bc * Sc;   // 4096
  float*  Qf  = (float*) alloc(M * Hc  * HDc * sizeof(float));
  float*  Kf  = (float*) alloc(M * KVc * HDc * sizeof(float));
  float*  Vf  = (float*) alloc(M * KVc * HDc * sizeof(float));
  __bf16* qb  = (__bf16*)alloc(M * Hc  * HDc * 2);
  __bf16* kb  = (__bf16*)alloc(M * KVc * HDc * 2);
  __bf16* vbT = (__bf16*)alloc(M * KVc * HDc * 2);
  float*  atn = (float*) alloc(M * Hc  * HDc * sizeof(float));
  float*  dcm = (float*) alloc((size_t)Bc * KVc * Sc * sizeof(float));

  dim3 blk(256);
  // QKV projections
  gemm_xw_bf16<Hc * HDc, HIDc><<<dim3((Hc * HDc) / BNg,  (unsigned)(M / BMg)),
      blk, 0, stream>>>(x, Wq, Qf);
  gemm_xw_bf16<KVc * HDc, HIDc><<<dim3((KVc * HDc) / BNg, (unsigned)(M / BMg)),
      blk, 0, stream>>>(x, Wk, Kf);
  gemm_xw_bf16<KVc * HDc, HIDc><<<dim3((KVc * HDc) / BNg, (unsigned)(M / BMg)),
      blk, 0, stream>>>(x, Wv, Vf);
  // RMS-norm + decisions + RoPE + bf16 pack (Q pre-scaled, V transposed)
  norm_rope_kernel<<<dim3((unsigned)(M * 32 / 8)), blk, 0, stream>>>(
      Qf, Kf, Vf, cosT, sinT, qn_w, kn_w, qb, kb, vbT, dcm);
  // Flash attention with DMS mask
  attn_kernel<<<dim3(Bc * Hc * (Sc / 128)), blk, 0, stream>>>(qb, kb, vbT, dcm, atn);
  // Output projection
  gemm_xw_bf16<HIDc, Hc * HDc><<<dim3(HIDc / BNg, (unsigned)(M / BMg)),
      blk, 0, stream>>>(atn, Wo, out);
}
